// CausalSelfAttention_30958124270153
// MI455X (gfx1250) — matrix operand order repaired and checked by reading, not compiled, not run on hardware
//
#include <hip/hip_runtime.h>
#include <hip/hip_bf16.h>
#include <cstdint>

typedef __attribute__((ext_vector_type(16))) _Float16 v16h;
typedef __attribute__((ext_vector_type(8)))  _Float16 v8h;
typedef __attribute__((ext_vector_type(8)))  float    v8f;

typedef __attribute__((ext_vector_type(4))) unsigned tdm_u32x4;
typedef __attribute__((ext_vector_type(8))) int      tdm_i32x8;
typedef __attribute__((ext_vector_type(4))) int      tdm_i32x4;

#define B_  2
#define T_  2048
#define C_  1024
#define H_  16
#define HD_ 64

union V16U { v16h v; v8h h[2]; _Float16 e[16]; };

// ---- CDNA5 async global->LDS copy (ASYNCcnt-tracked) ----------------------
__device__ __forceinline__ void async_b128(unsigned lds_addr, const void* gaddr) {
  asm volatile("global_load_async_to_lds_b128 %0, %1, off"
               :: "v"(lds_addr), "v"(gaddr) : "memory");
}
__device__ __forceinline__ void wait_async0() {
  asm volatile("s_wait_asynccnt 0x0" ::: "memory");
}
__device__ __forceinline__ unsigned lds_lo32(const void* p) {
  // flat LDS pointers keep the byte offset in addr[31:0] (aperture is [63:32])
  return (unsigned)(uintptr_t)p;
}

// ---- CDNA5 Tensor Data Mover: 2D tile global->LDS with LDS row padding ----
// Descriptor bit layout per CDNA5 ISA sec 8.3 (group0) / 8.4 (group1).
// tile_w, tile_h, row_stride in elements (data_size = 2 bytes).
// pad_interval code: 0=2,1=4,2=8,3=16,4=32,... DWORDs; pad_amount code: n-1 DWORDs.
__device__ __forceinline__ void tdm_load_2d(unsigned lds_addr, const void* gaddr,
                                            unsigned tile_w, unsigned tile_h,
                                            unsigned row_stride,
                                            unsigned pad_interval, unsigned pad_amount) {
  uint64_t ga = (uint64_t)gaddr;
  tdm_u32x4 g0;
  g0[0] = 1u;                                            // count=1, user D#
  g0[1] = lds_addr;                                      // lds_addr [63:32]
  g0[2] = (unsigned)ga;                                  // global_addr [95:64]
  g0[3] = (unsigned)((ga >> 32) & 0x01FFFFFFu)           // global_addr [120:96]
        | (2u << 30);                                    // type=2 ("image")
  tdm_i32x8 g1;
  g1[0] = (int)((1u << 16)                               // data_size = 2 bytes
              | (1u << 20)                               // pad_enable
              | (pad_interval << 22) | (pad_amount << 25));
  g1[1] = (int)(tile_w << 16);                           // tensor_dim0 lo16
  g1[2] = (int)(tile_h << 16);                           // td0 hi=0 | tensor_dim1 lo16
  g1[3] = (int)(tile_w << 16);                           // td1 hi=0 | tile_dim0
  g1[4] = (int)tile_h;                                   // tile_dim1 | tile_dim2=0
  g1[5] = (int)row_stride;                               // tensor_dim0_stride lo32
  g1[6] = 0;                                             // stride hi | dim1_stride lo
  g1[7] = 0;
  tdm_i32x4 z4 = {0, 0, 0, 0};
#if defined(__clang_major__) && __clang_major__ >= 23
  tdm_i32x8 z8 = {0, 0, 0, 0, 0, 0, 0, 0};
  __builtin_amdgcn_tensor_load_to_lds(g0, g1, z4, z4, z8, 0);
#else
  __builtin_amdgcn_tensor_load_to_lds(g0, g1, z4, z4, 0);
#endif
}
__device__ __forceinline__ void wait_tensor0() {
  __builtin_amdgcn_s_wait_tensorcnt((short)0);
}

// ---------------------------------------------------------------------------
// Pre-conversion helpers (run once; off the critical loop)
// ---------------------------------------------------------------------------
__global__ __launch_bounds__(256)
void cvt_f16(const float* __restrict__ in, _Float16* __restrict__ out, int n) {
  int i = blockIdx.x * 256 + threadIdx.x;
  if (i < n) out[i] = (_Float16)in[i];
}

__global__ __launch_bounds__(256)
void transpose_cvt(const float* __restrict__ W, _Float16* __restrict__ Wt,
                   int K, int N) {   // W[K][N] f32 -> Wt[N][K] f16
  size_t i = (size_t)blockIdx.x * 256 + threadIdx.x;
  if (i < (size_t)K * N) {
    int n = (int)(i % N), k = (int)(i / N);
    Wt[(size_t)n * K + k] = (_Float16)W[i];
  }
}

// qkv [B*T][3C] -> vt [B][H][hd][T]  (K-major V for PV B-fragments)
__global__ __launch_bounds__(256)
void transpose_v(const _Float16* __restrict__ qkv, _Float16* __restrict__ vt) {
  size_t i = (size_t)blockIdx.x * 256 + threadIdx.x;   // over B*H*HD*T
  int t  = (int)(i % T_);
  int r  = (int)(i / T_);          // (b*H + h)*64 + d
  int d  = r & 63;
  int hh = (r >> 6) & 15;
  int bb = r >> 10;
  vt[i] = qkv[((size_t)bb * T_ + t) * (3 * C_) + 2 * C_ + hh * 64 + d];
}

// ---------------------------------------------------------------------------
// WMMA GEMM, all-f16 operands, B pre-transposed: C[M,N] = A[M,K] * Bt[N,K]^T
// block tile 128x128, 256 threads (8 waves: 4 in M x 2 in N), wave tile 32x64
// -> 8 v_wmma per 32-wide K step. Double-buffered LDS fed by TDM tile loads.
// ---------------------------------------------------------------------------
template <typename OT>
__global__ __launch_bounds__(256)
void gemm_wmma_f16(const _Float16* __restrict__ A, const _Float16* __restrict__ Bt,
                   OT* __restrict__ Cm, int M, int N, int K) {
  constexpr int AS = 40;  // A row stride (f16): 64B data + 16B pad (4 DW)
  constexpr int BS = 48;  // B row stride (f16): 64B data + 32B pad (8 DW)
  constexpr int STG = 128 * AS + 128 * BS;               // f16 per stage
  __shared__ __align__(32) _Float16 smem[2 * STG];

  const int tid  = threadIdx.x;
  const int lane = tid & 31;
  const int wave = tid >> 5;
  const int hf   = lane >> 4;
  const int l16  = lane & 15;
  const int wm   = wave & 3;       // 0..3 -> M
  const int wn   = wave >> 2;      // 0..1 -> N
  const int M0   = blockIdx.y * 128;
  const int N0   = blockIdx.x * 128;

  const unsigned ldsBase = lds_lo32(smem);

  auto stage = [&](int k0, int buf) {    // wave 0 only; TDM ignores EXEC
    unsigned ab = ldsBase + (unsigned)buf * (STG * 2);
    unsigned bb = ab + 128 * AS * 2;
    tdm_load_2d(ab, A  + (size_t)M0 * K + k0, 32, 128, (unsigned)K, 3, 3);
    tdm_load_2d(bb, Bt + (size_t)N0 * K + k0, 32, 128, (unsigned)K, 3, 7);
  };

  v8f acc[2][4] = {};
  const int nsteps = K >> 5;
  if (wave == 0) stage(0, 0);

  for (int s = 0; s < nsteps; ++s) {
    if (wave == 0) wait_tensor0();   // current buffer's tiles landed
    __syncthreads();                 // publish; everyone done with next buffer
    if (wave == 0 && s + 1 < nsteps) stage((s + 1) << 5, (s + 1) & 1);

    const _Float16* Ab = smem + (size_t)(s & 1) * STG;
    const _Float16* Bb = Ab + 128 * AS;

    v16h af[2], bf[4];
    for (int mf = 0; mf < 2; ++mf) {             // A frag: k = 16*hf + i
      const _Float16* p = Ab + (wm * 32 + mf * 16 + l16) * AS;
      V16U u;
      u.h[0] = *(const v8h*)(p + 16 * hf);
      u.h[1] = *(const v8h*)(p + 16 * hf + 8);
      af[mf] = u.v;
    }
    for (int nf = 0; nf < 4; ++nf)               // B frag: k = 16*hf + i
      bf[nf] = *(const v16h*)(Bb + (wn * 64 + nf * 16 + l16) * BS + 16 * hf);

    for (int mf = 0; mf < 2; ++mf)
      for (int nf = 0; nf < 4; ++nf)
        acc[mf][nf] = __builtin_amdgcn_wmma_f32_16x16x32_f16(
            false, af[mf], false, bf[nf], (short)0, acc[mf][nf], false, false);
  }

  for (int mf = 0; mf < 2; ++mf)
    for (int nf = 0; nf < 4; ++nf)
      for (int j = 0; j < 8; ++j) {
        int row = M0 + wm * 32 + mf * 16 + j + 8 * hf;
        int col = N0 + wn * 64 + nf * 16 + l16;
        Cm[(size_t)row * N + col] = (OT)acc[mf][nf][j];
      }
}

// ---------------------------------------------------------------------------
// Causal flash attention. Grid (T/64, H, B), 128 threads (4 waves x 16 rows).
// qkv: [B*T][3C] f16 (Q at h*64, K at C+h*64); vt: [B][H][hd][T] f16.
// Q staged via ASYNC loads; K / V^T tiles staged via TDM per key block.
// ---------------------------------------------------------------------------
__global__ __launch_bounds__(128)
void attn_flash(const _Float16* __restrict__ qkv, const _Float16* __restrict__ vt,
                _Float16* __restrict__ out) {
  constexpr int QS = 80, KS = 80, VS = 48, PS = 32;
  __shared__ __align__(32) _Float16 Qs[64 * QS];       // [q][hd]
  __shared__ __align__(32) _Float16 Ks[32 * KS];       // [key][hd]
  __shared__ __align__(32) _Float16 Vs[64 * VS];       // [hd][key]
  __shared__ __align__(32) _Float16 Ps[4][16 * PS];    // per-wave P tile

  const int tid  = threadIdx.x;
  const int lane = tid & 31;
  const int wave = tid >> 5;
  const int hf   = lane >> 4;
  const int l16  = lane & 15;
  const int qb   = blockIdx.x * 64;
  const int h    = blockIdx.y;
  const int b    = blockIdx.z;

  const size_t base = (size_t)b * T_ * (3 * C_) + (size_t)h * HD_;
  const _Float16* vhead = vt + (size_t)(b * H_ + h) * HD_ * T_;

  const unsigned qsB = lds_lo32(Qs);
  const unsigned ksB = lds_lo32(Ks);
  const unsigned vsB = lds_lo32(Vs);

  // stage Q tile (64 rows x 8 chunks of 8 f16) via async copies
  for (int i = tid; i < 512; i += 128) {
    int r = i >> 3, c = (i & 7) * 8;
    async_b128(qsB + (unsigned)(r * QS + c) * 2,
               qkv + base + (size_t)(qb + r) * (3 * C_) + c);
  }
  wait_async0();
  __syncthreads();

  // Q A-fragments, constant across the key loop
  v16h qa[2];
  for (int kc = 0; kc < 2; ++kc) {
    const _Float16* p = &Qs[(wave * 16 + l16) * QS + kc * 32];
    V16U u;
    u.h[0] = *(const v8h*)(p + 16 * hf);
    u.h[1] = *(const v8h*)(p + 16 * hf + 8);
    qa[kc] = u.v;
  }

  float m[8], l[8];
  v8f o[4] = {};
  for (int j = 0; j < 8; ++j) { m[j] = -1e30f; l[j] = 0.0f; }

  const int   nkb   = qb / 32 + 2;
  const float scale = 0.125f;        // 1/sqrt(64)

  for (int kb = 0; kb < nkb; ++kb) {
    // stage K (32x64) and V^T (64x32) tiles via TDM (wave 0 issues)
    if (wave == 0) {
      tdm_load_2d(ksB, qkv + base + (size_t)(kb * 32) * (3 * C_) + C_,
                  64, 32, 3 * C_, 4, 7);
      tdm_load_2d(vsB, vhead + kb * 32,
                  32, 64, T_, 3, 7);
      wait_tensor0();
    }
    __syncthreads();

    // S = Q K^T : 16 q-rows x 32 keys
    v8f s[2];
    for (int g = 0; g < 2; ++g) {
      v8f a = {};
      for (int kc = 0; kc < 2; ++kc) {
        v16h kf = *(const v16h*)&Ks[(g * 16 + l16) * KS + kc * 32 + 16 * hf];
        a = __builtin_amdgcn_wmma_f32_16x16x32_f16(false, qa[kc], false, kf,
                                                   (short)0, a, false, false);
      }
      s[g] = a;
    }

    // mask + scale + online softmax (rows live in 16-lane halves)
    for (int j = 0; j < 8; ++j) {
      int qrow = qb + wave * 16 + j + 8 * hf;
      float v0 = (kb * 32 + l16      <= qrow) ? s[0][j] * scale : -1e30f;
      float v1 = (kb * 32 + 16 + l16 <= qrow) ? s[1][j] * scale : -1e30f;
      float mx = fmaxf(v0, v1);
      mx = fmaxf(mx, __shfl_xor(mx, 1));
      mx = fmaxf(mx, __shfl_xor(mx, 2));
      mx = fmaxf(mx, __shfl_xor(mx, 4));
      mx = fmaxf(mx, __shfl_xor(mx, 8));
      float mn    = fmaxf(m[j], mx);
      float alpha = __expf(m[j] - mn);
      float p0 = __expf(v0 - mn);
      float p1 = __expf(v1 - mn);
      s[0][j] = p0; s[1][j] = p1;
      float ps = p0 + p1;
      ps += __shfl_xor(ps, 1);
      ps += __shfl_xor(ps, 2);
      ps += __shfl_xor(ps, 4);
      ps += __shfl_xor(ps, 8);
      l[j] = l[j] * alpha + ps;
      m[j] = mn;
      for (int d = 0; d < 4; ++d) o[d][j] *= alpha;
    }

    // restage P (C-layout) into A-layout via per-wave LDS tile
    for (int g = 0; g < 2; ++g)
      for (int j = 0; j < 8; ++j)
        Ps[wave][(j + 8 * hf) * PS + g * 16 + l16] = (_Float16)s[g][j];
    asm volatile("s_wait_dscnt 0x0" ::: "memory");

    V16U u;
    const _Float16* pp = &Ps[wave][l16 * PS];
    u.h[0] = *(const v8h*)(pp + 16 * hf);
    u.h[1] = *(const v8h*)(pp + 16 * hf + 8);
    v16h pa = u.v;

    // O += P * V
    for (int d = 0; d < 4; ++d) {
      v16h vf = *(const v16h*)&Vs[(d * 16 + l16) * VS + 16 * hf];
      o[d] = __builtin_amdgcn_wmma_f32_16x16x32_f16(false, pa, false, vf,
                                                    (short)0, o[d], false, false);
    }
    __syncthreads();   // protect Ks/Vs before next iteration's TDM issue
  }

  // normalize and write merged-head output [B,T,C] f16
  for (int j = 0; j < 8; ++j) {
    float inv = 1.0f / l[j];
    int t = qb + wave * 16 + j + 8 * hf;
    size_t ro = ((size_t)b * T_ + t) * C_ + (size_t)h * HD_;
    for (int d = 0; d < 4; ++d)
      out[ro + d * 16 + l16] = (_Float16)(o[d][j] * inv);
  }
}

// ---------------------------------------------------------------------------
extern "C" void kernel_launch(void* const* d_in, const int* in_sizes, int n_in,
                              void* d_out, int out_size, void* d_ws, size_t ws_size,
                              hipStream_t stream) {
  (void)in_sizes; (void)n_in; (void)out_size; (void)ws_size;
  const float* x    = (const float*)d_in[0];   // [B,T,C]
  const float* Wqkv = (const float*)d_in[1];   // [C,3C]
  const float* Wout = (const float*)d_in[2];   // [C,C]
  float*       out  = (float*)d_out;           // [B,T,C]

  const size_t BT = (size_t)B_ * T_;
  _Float16* x16   = (_Float16*)d_ws;                 // [B*T][C]
  _Float16* wqkvT = x16   + BT * C_;                 // [3C][C]
  _Float16* woutT = wqkvT + (size_t)3 * C_ * C_;     // [C][C]
  _Float16* qkv   = woutT + (size_t)C_ * C_;         // [B*T][3C]
  _Float16* vt    = qkv   + BT * 3 * C_;             // [B][H][hd][T]
  _Float16* attn  = vt    + BT * C_;                 // [B*T][C]

  // 0) one-time conversions / transposes to f16
  cvt_f16<<<(int)(BT * C_ / 256), 256, 0, stream>>>(x, x16, (int)(BT * C_));
  transpose_cvt<<<(int)((size_t)C_ * 3 * C_ / 256), 256, 0, stream>>>(
      Wqkv, wqkvT, C_, 3 * C_);
  transpose_cvt<<<(int)((size_t)C_ * C_ / 256), 256, 0, stream>>>(
      Wout, woutT, C_, C_);

  // 1) QKV projection (f16 WMMA, TDM-staged, double-buffered)
  dim3 g1((3 * C_) / 128, (int)(BT / 128));
  gemm_wmma_f16<_Float16><<<g1, 256, 0, stream>>>(x16, wqkvT, qkv,
                                                  (int)BT, 3 * C_, C_);

  // 2) V -> V^T per head, then causal flash attention
  transpose_v<<<(int)(BT * C_ / 256), 256, 0, stream>>>(qkv, vt);
  dim3 g2(T_ / 64, H_, B_);
  attn_flash<<<g2, 128, 0, stream>>>(qkv, vt, attn);

  // 3) output projection (f32 result)
  dim3 g3(C_ / 128, (int)(BT / 128));
  gemm_wmma_f16<float><<<g3, 256, 0, stream>>>(attn, woutT, out,
                                               (int)BT, C_, C_);
}
